// BidirectMambaReg_30691836297500
// MI455X (gfx1250) — compile-verified, matrix operand order checked by reading, not run on hardware
//
#include <hip/hip_runtime.h>

namespace {

constexpr int B_   = 32;
constexpr int L_   = 512;
constexpr int HID_ = 256;
constexpr int ED_  = 512;
constexpr int N_   = 16;
constexpr int BL   = B_ * L_;       // 16384 rows
constexpr int PROJC = 4 * ED_;      // 2048

typedef __attribute__((ext_vector_type(16))) _Float16 v16h;
typedef __attribute__((ext_vector_type(8)))  _Float16 v8h;
typedef __attribute__((ext_vector_type(8)))  float    v8f;

__device__ __forceinline__ float silu_f(float x) { return x / (1.0f + __expf(-x)); }

// Load one 16-half WMMA fragment (CDNA5 16-bit 16x32 striping: the caller's
// pointer is already offset by lane-dependent khalf; the two halves sit 16
// elements apart). Lowers to two global_load_b128 into consecutive VGPRs.
__device__ __forceinline__ v16h load_frag(const _Float16* __restrict__ p)
{
    const v8h lo = *(const v8h*)(p);
    const v8h hi = *(const v8h*)(p + 16);
    v16h f;
#pragma unroll
    for (int i = 0; i < 8; ++i) { f[i] = lo[i]; f[8 + i] = hi[i]; }
    return f;
}

// ---------------------------------------------------------------------------
// WMMA GEMM, fully static tiling: one wave computes (MT*16) x (NT*16) of
// C[M,N] = act(A[M,K] @ Bt[N,K]^T + bias + resid).  K known at compile time;
// double-buffered fragments so next chunk's loads are in flight during the
// current chunk's WMMAs.
// ---------------------------------------------------------------------------
template <int MT, int NT, int K>
__global__ __launch_bounds__(32)
void k_gemm(const _Float16* __restrict__ A,
            const _Float16* __restrict__ Bt,
            const float*    __restrict__ bias,
            const float*    __restrict__ resid,
            float*          __restrict__ Cf,
            _Float16*       __restrict__ Ch,
            int N, int act)
{
    static_assert(K % 64 == 0, "K must be a multiple of 64");
    const int lane  = threadIdx.x & 31;
    const int mrow  = lane & 15;
    const int khalf = (lane >> 4) * 8;          // 0 or 8 (K striping)
    const int mbase = blockIdx.x * (MT * 16);
    const int nbase = blockIdx.y * (NT * 16);

    const _Float16* aRow[MT];
    const _Float16* bRow[NT];
#pragma unroll
    for (int mt = 0; mt < MT; ++mt)
        aRow[mt] = A + (size_t)(mbase + mt * 16 + mrow) * K + khalf;
#pragma unroll
    for (int nt = 0; nt < NT; ++nt)
        bRow[nt] = Bt + (size_t)(nbase + nt * 16 + mrow) * K + khalf;

    v8f acc[MT][NT];
#pragma unroll
    for (int mt = 0; mt < MT; ++mt)
#pragma unroll
        for (int nt = 0; nt < NT; ++nt)
#pragma unroll
            for (int r = 0; r < 8; ++r) acc[mt][nt][r] = 0.0f;

    v16h a0[MT], a1[MT], b0[NT], b1[NT];
#pragma unroll
    for (int mt = 0; mt < MT; ++mt) a0[mt] = load_frag(aRow[mt]);
#pragma unroll
    for (int nt = 0; nt < NT; ++nt) b0[nt] = load_frag(bRow[nt]);

    for (int k0 = 0; k0 < K; k0 += 64) {
        // issue loads for chunk k0+32 before consuming chunk k0
#pragma unroll
        for (int mt = 0; mt < MT; ++mt) a1[mt] = load_frag(aRow[mt] + k0 + 32);
#pragma unroll
        for (int nt = 0; nt < NT; ++nt) b1[nt] = load_frag(bRow[nt] + k0 + 32);
#pragma unroll
        for (int mt = 0; mt < MT; ++mt)
#pragma unroll
            for (int nt = 0; nt < NT; ++nt)
                acc[mt][nt] = __builtin_amdgcn_wmma_f32_16x16x32_f16(
                    false, a0[mt], false, b0[nt], (short)0, acc[mt][nt], false, false);

        if (k0 + 64 < K) {
#pragma unroll
            for (int mt = 0; mt < MT; ++mt) a0[mt] = load_frag(aRow[mt] + k0 + 64);
#pragma unroll
            for (int nt = 0; nt < NT; ++nt) b0[nt] = load_frag(bRow[nt] + k0 + 64);
        }
#pragma unroll
        for (int mt = 0; mt < MT; ++mt)
#pragma unroll
            for (int nt = 0; nt < NT; ++nt)
                acc[mt][nt] = __builtin_amdgcn_wmma_f32_16x16x32_f16(
                    false, a1[mt], false, b1[nt], (short)0, acc[mt][nt], false, false);
    }

    // epilogue (runs once; runtime branches here are cheap)
    const int ncol = lane & 15;
    const int rofs = (lane >> 4) * 8;
#pragma unroll
    for (int mt = 0; mt < MT; ++mt)
#pragma unroll
        for (int nt = 0; nt < NT; ++nt) {
            const int n = nbase + nt * 16 + ncol;
            const float bv = bias ? bias[n] : 0.0f;
#pragma unroll
            for (int r = 0; r < 8; ++r) {
                const int m = mbase + mt * 16 + rofs + r;
                float v = acc[mt][nt][r] + bv;
                if (resid) v += resid[(size_t)m * N + n];
                if (act == 1) v = fmaxf(v, 0.0f);
                if (Cf) Cf[(size_t)m * N + n] = v;
                if (Ch) Ch[(size_t)m * N + n] = (_Float16)v;
            }
        }
}

// ---------------------------------------------------------------------------
// Weight prep
// ---------------------------------------------------------------------------
__global__ void k_transpose_h(const float* __restrict__ W, _Float16* __restrict__ Wt,
                              int Kdim, int Ndim)
{
    size_t i = (size_t)blockIdx.x * 256 + threadIdx.x;
    if (i >= (size_t)Kdim * Ndim) return;
    int n = (int)(i % Ndim), k = (int)(i / Ndim);
    Wt[(size_t)n * Kdim + k] = (_Float16)W[i];
}

__global__ void k_pack_bcdw(const float* __restrict__ Bw, const float* __restrict__ Cw,
                            const float* __restrict__ d1w, _Float16* __restrict__ out)
{
    int i = blockIdx.x * 256 + threadIdx.x;          // 48*512
    if (i >= 48 * ED_) return;
    int k = i % ED_, row = i / ED_;
    const float* src = (row < 16) ? Bw : (row < 32 ? Cw : d1w);
    out[i] = (_Float16)src[(size_t)k * 16 + (row & 15)];
}

__global__ void k_pack_bcdb(const float* Bb, const float* Cb, const float* d1b, float* out)
{
    int i = threadIdx.x;                             // 48
    out[i] = (i < 16) ? Bb[i] : (i < 32 ? Cb[i - 16] : d1b[i - 32]);
}

// ---------------------------------------------------------------------------
// Embedding gather with pad mask
// ---------------------------------------------------------------------------
__global__ void k_embed(const int* __restrict__ seq_c, const float* __restrict__ emb,
                        float* __restrict__ x)
{
    size_t i = (size_t)blockIdx.x * 256 + threadIdx.x;   // B*L*HID
    int h = (int)(i % HID_);
    size_t bt = i / HID_;
    int t = (int)(bt % L_), b = (int)(bt / L_);
    int tok = seq_c[(size_t)b * (L_ + 1) + t];
    x[i] = (tok != 0) ? emb[(size_t)tok * HID_ + h] : 0.0f;
}

// ---------------------------------------------------------------------------
// RMSNorm: one 256-thread block per (b,t) row; f16 output for WMMA input.
// ---------------------------------------------------------------------------
__global__ void k_rmsnorm(const float* __restrict__ x, const float* __restrict__ scale,
                          _Float16* __restrict__ xn_h)
{
    __shared__ float red[256];
    const size_t row = blockIdx.x;
    const float v = x[row * HID_ + threadIdx.x];
    red[threadIdx.x] = v * v;
    __syncthreads();
    for (int s = 128; s > 0; s >>= 1) {
        if ((int)threadIdx.x < s) red[threadIdx.x] += red[threadIdx.x + s];
        __syncthreads();
    }
    const float inv = rsqrtf(red[0] * (1.0f / HID_) + 1e-6f);
    xn_h[row * HID_ + threadIdx.x] = (_Float16)(v * inv * scale[threadIdx.x]);
}

// ---------------------------------------------------------------------------
// Causal depthwise conv (K=3) + SiLU, both directions in one launch.
// u_h layout: [dir][b][s][d], s = scan-time (reversed for dir=1).
// ---------------------------------------------------------------------------
__global__ void k_conv_silu(const float* __restrict__ proj,
                            const float* __restrict__ cwf, const float* __restrict__ cbf,
                            const float* __restrict__ cwr, const float* __restrict__ cbr,
                            _Float16* __restrict__ u_h)
{
    size_t i = (size_t)blockIdx.x * 256 + threadIdx.x;   // 2*B*L*ED
    int d = (int)(i % ED_);
    size_t tmp = i / ED_;
    int s = (int)(tmp % L_);
    tmp /= L_;
    int b = (int)(tmp % B_);
    int dir = (int)(tmp / B_);
    const float* w  = dir ? cwr : cwf;
    const float* bb = dir ? cbr : cbf;
    float acc = bb[d];
#pragma unroll
    for (int j = 0; j < 3; ++j) {
        int ss = s - 2 + j;
        if (ss >= 0) {
            int t = dir ? (L_ - 1 - ss) : ss;
            acc += proj[((size_t)b * L_ + t) * PROJC + dir * ED_ + d] * w[j * ED_ + d];
        }
    }
    u_h[i] = (_Float16)silu_f(acc);
}

// ---------------------------------------------------------------------------
// Selective scan, fused with dt=softplus(dtpre@dt2w+dt2b) and output gating
// y * silu(z). One thread per (dir,b,d); 16 states in registers; per-step
// Bm/Cm/dtpre (48 floats) staged in LDS.
// ---------------------------------------------------------------------------
__global__ __launch_bounds__(256)
void k_scan(const _Float16* __restrict__ u_h, const float* __restrict__ bcd,
            const float* __restrict__ proj,
            const float* __restrict__ dt2w_f, const float* __restrict__ dt2b_f,
            const float* __restrict__ Alog_f, const float* __restrict__ D_f,
            const float* __restrict__ dt2w_r, const float* __restrict__ dt2b_r,
            const float* __restrict__ Alog_r, const float* __restrict__ D_r,
            float* __restrict__ gated, _Float16* __restrict__ gated_h)
{
    const int b   = blockIdx.x % B_;
    const int dir = blockIdx.x / B_;
    const int d   = blockIdx.y * 256 + threadIdx.x;

    const float* dt2w = dir ? dt2w_r : dt2w_f;
    const float* dt2b = dir ? dt2b_r : dt2b_f;
    const float* Alog = dir ? Alog_r : Alog_f;
    const float* Dv   = dir ? D_r    : D_f;

    float Arow[16], W2[16], h[16];
#pragma unroll
    for (int n = 0; n < 16; ++n) {
        Arow[n] = -__expf(Alog[(size_t)d * N_ + n]);
        W2[n]   = dt2w[(size_t)n * ED_ + d];
        h[n]    = 0.0f;
    }
    const float dtb = dt2b[d];
    const float Dd  = Dv[d];

    __shared__ float sh[48];
    const size_t ub = ((size_t)dir * B_ + b) * (size_t)L_ * ED_;
    const size_t cb = ((size_t)dir * B_ + b) * (size_t)L_ * 48;

    for (int s = 0; s < L_; ++s) {
        if (threadIdx.x < 48) sh[threadIdx.x] = bcd[cb + (size_t)s * 48 + threadIdx.x];
        __syncthreads();

        float dtp = dtb;
#pragma unroll
        for (int r = 0; r < 16; ++r) dtp += sh[32 + r] * W2[r];
        const float dt = (dtp > 20.0f) ? dtp : log1pf(__expf(dtp));
        const float u  = (float)u_h[ub + (size_t)s * ED_ + d];
        const float du = dt * u;

        float y = 0.0f;
#pragma unroll
        for (int n = 0; n < 16; ++n) {
            h[n] = __expf(dt * Arow[n]) * h[n] + du * sh[n];
            y += h[n] * sh[16 + n];
        }
        y += Dd * u;

        const int t = dir ? (L_ - 1 - s) : s;
        const float z = proj[((size_t)b * L_ + t) * PROJC + 2 * ED_ + dir * ED_ + d];
        const float g = y * silu_f(z);
        const size_t go = ((size_t)b * L_ + t) * (size_t)(2 * ED_) + dir * ED_ + d;
        gated[go]   = g;
        gated_h[go] = (_Float16)g;
        __syncthreads();
    }
}

// h = concat(xfg[:, L-1, :], xrg[:, 0, :])  -> (B, 1024) f16
__global__ void k_gather_h(const float* __restrict__ gated, _Float16* __restrict__ h_h)
{
    int i = blockIdx.x * 256 + threadIdx.x;     // B*1024
    int j = i % 1024, b = i / 1024;
    const float v = (j < ED_) ? gated[((size_t)b * L_ + (L_ - 1)) * 1024 + j]
                              : gated[((size_t)b * L_ + 0) * 1024 + j];
    h_h[i] = (_Float16)v;
}

// fit_coeff = exp(h2 @ fit_w + fit_b); out = [fc, fc*c]
__global__ void k_final(const float* __restrict__ h2, const float* __restrict__ fit_w,
                        const float* __restrict__ fit_b, const int* __restrict__ seq_c,
                        float* __restrict__ out)
{
    const int b = threadIdx.x;                  // 32
    float acc = fit_b[0];
    for (int k = 0; k < 256; ++k) acc += h2[(size_t)b * 256 + k] * fit_w[k];
    const float fc = __expf(acc);
    const float c  = (float)seq_c[(size_t)b * (L_ + 1) + L_];
    out[b]       = fc;
    out[32 + b]  = fc * c;
}

} // namespace

// ---------------------------------------------------------------------------
extern "C" void kernel_launch(void* const* d_in, const int* in_sizes, int n_in,
                              void* d_out, int out_size, void* d_ws, size_t ws_size,
                              hipStream_t stream)
{
    (void)in_sizes; (void)n_in; (void)out_size; (void)ws_size;

    auto F = [&](int i) { return (const float*)d_in[i]; };
    const int* seq_c   = (const int*)d_in[0];
    const float* embed = F(1);

    // flat input index map (insertion-order recursive flatten)
    const int blkBase[2] = {2, 31};   // rms,in_w,in_b, fw[12], rv[12], (out_w,out_b)
    struct SSM { const float *conv_w,*conv_b,*Bw,*Bb,*Cw,*Cb,*dt1w,*dt1b,*dt2w,*dt2b,*A_log,*D; };
    auto getSSM = [&](int a) {
        SSM s{F(a+0),F(a+1),F(a+2),F(a+3),F(a+4),F(a+5),F(a+6),F(a+7),F(a+8),F(a+9),F(a+10),F(a+11)};
        return s;
    };

    // workspace bump allocator (256-B aligned)
    char* wp = (char*)d_ws;
    auto alloc = [&](size_t bytes) -> void* {
        void* r = (void*)wp;
        wp += (bytes + 255) & ~(size_t)255;
        return r;
    };
    float*     x       = (float*)alloc((size_t)BL * HID_ * 4);
    _Float16*  xn_h    = (_Float16*)alloc((size_t)BL * HID_ * 2);
    float*     proj    = (float*)alloc((size_t)BL * PROJC * 4);
    _Float16*  u_h     = (_Float16*)alloc((size_t)2 * BL * ED_ * 2);
    float*     bcd     = (float*)alloc((size_t)2 * BL * 48 * 4);
    float*     gated   = (float*)alloc((size_t)BL * 2 * ED_ * 4);
    _Float16*  gated_h = (_Float16*)alloc((size_t)BL * 2 * ED_ * 2);
    _Float16*  in_wt[2]  = { (_Float16*)alloc((size_t)PROJC * HID_ * 2),
                             (_Float16*)alloc((size_t)PROJC * HID_ * 2) };
    _Float16*  out_wt    = (_Float16*)alloc((size_t)HID_ * 1024 * 2);
    _Float16*  bcdw[2][2]; float* bcdb[2][2];
    for (int i = 0; i < 2; ++i)
        for (int j = 0; j < 2; ++j) {
            bcdw[i][j] = (_Float16*)alloc((size_t)48 * ED_ * 2);
            bcdb[i][j] = (float*)alloc(48 * 4);
        }
    _Float16*  w1t  = (_Float16*)alloc((size_t)512 * 1024 * 2);
    _Float16*  w2t  = (_Float16*)alloc((size_t)256 * 512 * 2);
    _Float16*  h_h  = (_Float16*)alloc((size_t)B_ * 1024 * 2);
    _Float16*  h1_h = (_Float16*)alloc((size_t)B_ * 512 * 2);
    float*     h2   = (float*)alloc((size_t)B_ * 256 * 4);

    // ---- weight prep (f16 transposed copies + packed B/C/dt1) ----
    for (int blk = 0; blk < 2; ++blk) {
        const int base = blkBase[blk];
        k_transpose_h<<<(HID_ * PROJC) / 256, 256, 0, stream>>>(F(base + 1), in_wt[blk], HID_, PROJC);
        for (int dir = 0; dir < 2; ++dir) {
            SSM s = getSSM(base + 3 + dir * 12);
            k_pack_bcdw<<<(48 * ED_) / 256, 256, 0, stream>>>(s.Bw, s.Cw, s.dt1w, bcdw[blk][dir]);
            k_pack_bcdb<<<1, 48, 0, stream>>>(s.Bb, s.Cb, s.dt1b, bcdb[blk][dir]);
        }
    }
    k_transpose_h<<<(1024 * 256) / 256, 256, 0, stream>>>(F(29), out_wt, 1024, 256);
    k_transpose_h<<<(1024 * 512) / 256, 256, 0, stream>>>(F(58), w1t, 1024, 512);
    k_transpose_h<<<(512 * 256) / 256, 256, 0, stream>>>(F(60), w2t, 512, 256);

    // ---- forward ----
    k_embed<<<(BL * HID_) / 256, 256, 0, stream>>>(seq_c, embed, x);

    for (int blk = 0; blk < 2; ++blk) {
        const int base = blkBase[blk];
        SSM sf = getSSM(base + 3), sr = getSSM(base + 15);

        k_rmsnorm<<<BL, 256, 0, stream>>>(x, F(base + 0), xn_h);

        // in-proj: (16384 x 256) @ (256 x 2048)  -> WMMA, 32x64 tiles/wave
        k_gemm<2, 4, 256><<<dim3(BL / 32, PROJC / 64), 32, 0, stream>>>(
            xn_h, in_wt[blk], F(base + 2), nullptr, proj, nullptr, PROJC, 0);

        k_conv_silu<<<(2 * BL * ED_) / 256, 256, 0, stream>>>(
            proj, sf.conv_w, sf.conv_b, sr.conv_w, sr.conv_b, u_h);

        // fused B/C/dt1 projection per direction: (16384 x 512) @ (512 x 48)
        for (int dir = 0; dir < 2; ++dir) {
            k_gemm<2, 3, 512><<<dim3(BL / 32, 1), 32, 0, stream>>>(
                u_h + (size_t)dir * BL * ED_, bcdw[blk][dir], bcdb[blk][dir], nullptr,
                bcd + (size_t)dir * BL * 48, nullptr, 48, 0);
        }

        k_scan<<<dim3(2 * B_, ED_ / 256), 256, 0, stream>>>(
            u_h, bcd, proj,
            sf.dt2w, sf.dt2b, sf.A_log, sf.D,
            sr.dt2w, sr.dt2b, sr.A_log, sr.D,
            gated, gated_h);

        if (blk == 0) {
            // residual out-proj: x += gated @ out_w  (16384 x 1024 x 256)
            k_gemm<2, 4, 1024><<<dim3(BL / 32, HID_ / 64), 32, 0, stream>>>(
                gated_h, out_wt, F(30), x, x, nullptr, HID_, 0);
        }
    }

    k_gather_h<<<(B_ * 1024) / 256, 256, 0, stream>>>(gated, h_h);
    // head: (32 x 1024) @ (1024 x 512) relu ; (32 x 512) @ (512 x 256) relu
    k_gemm<2, 4, 1024><<<dim3(1, 512 / 64), 32, 0, stream>>>(
        h_h, w1t, F(59), nullptr, nullptr, h1_h, 512, 1);
    k_gemm<2, 4, 512><<<dim3(1, 256 / 64), 32, 0, stream>>>(
        h1_h, w2t, F(61), nullptr, h2, nullptr, 256, 1);
    k_final<<<1, 32, 0, stream>>>(h2, F(62), F(63), seq_c, (float*)d_out);
}
